// SelfAttention_62405874811167
// MI455X (gfx1250) — matrix operand order repaired and checked by reading, not compiled, not run on hardware
//
#include <hip/hip_runtime.h>
#include <hip/hip_bf16.h>
#include <stdint.h>

#define DIM 1536
#define NH  12
#define HD  128
#define SEQ 3072

typedef _Float16 v16h __attribute__((ext_vector_type(16)));
typedef float    v8f  __attribute__((ext_vector_type(8)));

union Frag {
    v16h     h;
    uint32_t u[8];
    uint4    q4[2];
    _Float16 e[16];
};

#if defined(__HIP_DEVICE_COMPILE__) && __has_builtin(__builtin_amdgcn_global_load_async_to_lds_b128)
#define ASYNC_LDS 1
typedef int v4i_vs __attribute__((vector_size(16)));
typedef __attribute__((address_space(1))) v4i_vs* gas_ptr_t;
typedef __attribute__((address_space(3))) v4i_vs* las_ptr_t;
#define GLB_PTR(p) ((gas_ptr_t)(p))
#define LDS_PTR(p) ((las_ptr_t)(p))
#endif

__device__ __forceinline__ void wait_async_lds() {
#if defined(__HIP_DEVICE_COMPILE__)
#if __has_builtin(__builtin_amdgcn_s_wait_asynccnt)
    __builtin_amdgcn_s_wait_asynccnt(0);
#else
    asm volatile("s_wait_asynccnt 0x0" ::: "memory");
#endif
#endif
}

// ---------------------------------------------------------------------------
// fp32 -> f16 elementwise
__global__ __launch_bounds__(256) void f32_to_f16_kernel(const float* __restrict__ in,
                                                         _Float16* __restrict__ out, int n) {
    int i = blockIdx.x * 256 + threadIdx.x;
    if (i < n) out[i] = (_Float16)in[i];
}

// W [K][N] fp32 -> Wt [N][K] f16 (LDS-tiled transpose)
__global__ __launch_bounds__(256) void transpose_f16_kernel(const float* __restrict__ W,
                                                            _Float16* __restrict__ Wt) {
    __shared__ float tile[32][33];
    int bx = blockIdx.x, by = blockIdx.y;
    int tx = threadIdx.x & 31, ty = threadIdx.x >> 5;   // 32 x 8
    for (int i = 0; i < 32; i += 8)
        tile[ty + i][tx] = W[(size_t)(by * 32 + ty + i) * DIM + bx * 32 + tx];
    __syncthreads();
    for (int i = 0; i < 32; i += 8)
        Wt[(size_t)(bx * 32 + ty + i) * DIM + by * 32 + tx] = (_Float16)tile[tx][ty + i];
}

// ---------------------------------------------------------------------------
// C[M][N] fp32 = A[M][K](f16) @ Bt[N][K](f16, row=N col=K) + bias[N]
// Block tile 128x128, 256 threads = 8 waves; double-buffered async LDS staging.
#define TK  32
#define PAD 8   // LDS row stride 40 halfs = 80B: keeps 16B alignment, spreads banks

__global__ __launch_bounds__(256) void gemm_bias_f16(const _Float16* __restrict__ A,
                                                     const _Float16* __restrict__ Bt,
                                                     const float* __restrict__ bias,
                                                     float* __restrict__ C,
                                                     int M, int N, int K) {
    __shared__ _Float16 As[2][128][TK + PAD];
    __shared__ _Float16 Bs[2][128][TK + PAD];
    const int m0 = blockIdx.y * 128;
    const int n0 = blockIdx.x * 128;
    const int tid = threadIdx.x;
    const int wave = tid >> 5, lane = tid & 31;
    const int half_ = lane >> 4, l16 = lane & 15;

    auto stage_tile = [&](int buf, int k0) {
#ifdef ASYNC_LDS
        #pragma unroll
        for (int i = 0; i < 2; ++i) {
            int idx = tid + i * 256;
            int r = idx >> 2, qd = idx & 3;
            __builtin_amdgcn_global_load_async_to_lds_b128(
                GLB_PTR(&A[(size_t)(m0 + r) * K + k0 + qd * 8]),
                LDS_PTR(&As[buf][r][qd * 8]), 0, 0);
            __builtin_amdgcn_global_load_async_to_lds_b128(
                GLB_PTR(&Bt[(size_t)(n0 + r) * K + k0 + qd * 8]),
                LDS_PTR(&Bs[buf][r][qd * 8]), 0, 0);
        }
#else
        uint4 ta[2], tb[2];
        #pragma unroll
        for (int i = 0; i < 2; ++i) {
            int idx = tid + i * 256;
            int r = idx >> 2, qd = idx & 3;
            ta[i] = *(const uint4*)&A[(size_t)(m0 + r) * K + k0 + qd * 8];
            tb[i] = *(const uint4*)&Bt[(size_t)(n0 + r) * K + k0 + qd * 8];
        }
        #pragma unroll
        for (int i = 0; i < 2; ++i) {
            int idx = tid + i * 256;
            int r = idx >> 2, qd = idx & 3;
            *(uint4*)&As[buf][r][qd * 8] = ta[i];
            *(uint4*)&Bs[buf][r][qd * 8] = tb[i];
        }
#endif
    };

    v8f acc[8];
    #pragma unroll
    for (int nf = 0; nf < 8; ++nf)
        for (int r = 0; r < 8; ++r) acc[nf][r] = 0.f;

    stage_tile(0, 0);
    wait_async_lds();
    __syncthreads();

    int cur = 0;
    for (int k0 = 0; k0 < K; k0 += TK) {
        // Prefetch next tile into the other buffer while we compute this one.
        if (k0 + TK < K) stage_tile(cur ^ 1, k0 + TK);

        // A fragment (16x32): lane row = l16, contiguous K: k = 16*half_ + element (same as B)
        Frag a;
        const int ar = wave * 16 + l16;
        const int abase = half_ * 16;
        #pragma unroll
        for (int p = 0; p < 8; ++p) {
            int kk = abase + p * 2;
            a.u[p] = *(const uint32_t*)&As[cur][ar][kk];
        }
        #pragma unroll
        for (int nf = 0; nf < 8; ++nf) {
            Frag b;
            const int br = nf * 16 + l16;
            const int bbase = half_ * 16;
            #pragma unroll
            for (int p = 0; p < 8; ++p)
                b.u[p] = *(const uint32_t*)&Bs[cur][br][bbase + p * 2];
            acc[nf] = __builtin_amdgcn_wmma_f32_16x16x32_f16(false, a.h, false, b.h,
                                                             (short)0, acc[nf], false, false);
        }

        wait_async_lds();   // my prefetch transfers have landed
        __syncthreads();    // everyone done reading cur + writing cur^1
        cur ^= 1;
    }

    #pragma unroll
    for (int nf = 0; nf < 8; ++nf)
        #pragma unroll
        for (int r = 0; r < 8; ++r) {
            int row = m0 + wave * 16 + r + half_ * 8;
            int col = n0 + nf * 16 + l16;
            C[(size_t)row * N + col] = acc[nf][r] + bias[col];
        }
}

// ---------------------------------------------------------------------------
// Per-token RMSNorm(q,k) + RoPE(q,k) + window-partition scatter.
// qh/kh: [dest][DIM] f16 ; vt: [DIM][dest] f16 (transposed for P@V B-fragments)
__device__ __forceinline__ int dest_of(int s) {
    int fi = s / 96, rem = s % 96, hi = rem / 12, wi = rem % 12;
    int b = ((fi >> 3) * 2 + (hi >> 2)) * 3 + (wi >> 2);
    int t = ((fi & 7) * 4 + (hi & 3)) * 4 + (wi & 3);
    return b * 128 + t;
}

__global__ __launch_bounds__(256) void norm_rope_scatter(const float* __restrict__ qp,
                                                         const float* __restrict__ kp,
                                                         const float* __restrict__ vp,
                                                         const float* __restrict__ freqs,
                                                         const float* __restrict__ gq,
                                                         const float* __restrict__ gk,
                                                         _Float16* __restrict__ qh,
                                                         _Float16* __restrict__ kh,
                                                         _Float16* __restrict__ vt) {
    const int s = blockIdx.x;           // original token index
    const int dest = dest_of(s);
    const int tid = threadIdx.x;
    __shared__ float red[256];

    const float* qrow = qp + (size_t)s * DIM;
    const float* krow = kp + (size_t)s * DIM;
    const float* vrow = vp + (size_t)s * DIM;

    float sq = 0.f, sk = 0.f;
    for (int i = tid; i < DIM; i += 256) {
        float a = qrow[i]; sq += a * a;
        float b = krow[i]; sk += b * b;
    }
    red[tid] = sq; __syncthreads();
    for (int off = 128; off > 0; off >>= 1) {
        if (tid < off) red[tid] += red[tid + off];
        __syncthreads();
    }
    float rq = rsqrtf(red[0] * (1.f / DIM) + 1e-5f);
    __syncthreads();
    red[tid] = sk; __syncthreads();
    for (int off = 128; off > 0; off >>= 1) {
        if (tid < off) red[tid] += red[tid + off];
        __syncthreads();
    }
    float rk = rsqrtf(red[0] * (1.f / DIM) + 1e-5f);

    // RoPE pairs: p = head*64 + d  ->  dims (2p, 2p+1)
    for (int p = tid; p < DIM / 2; p += 256) {
        int d = p & 63;
        float c  = freqs[(size_t)(s * 64 + d) * 2 + 0];
        float sn = freqs[(size_t)(s * 64 + d) * 2 + 1];
        float qa = qrow[2 * p]     * rq * gq[2 * p];
        float qb = qrow[2 * p + 1] * rq * gq[2 * p + 1];
        qh[(size_t)dest * DIM + 2 * p]     = (_Float16)(qa * c - qb * sn);
        qh[(size_t)dest * DIM + 2 * p + 1] = (_Float16)(qa * sn + qb * c);
        float ka = krow[2 * p]     * rk * gk[2 * p];
        float kb = krow[2 * p + 1] * rk * gk[2 * p + 1];
        kh[(size_t)dest * DIM + 2 * p]     = (_Float16)(ka * c - kb * sn);
        kh[(size_t)dest * DIM + 2 * p + 1] = (_Float16)(ka * sn + kb * c);
    }
    for (int i = tid; i < DIM; i += 256)
        vt[(size_t)i * SEQ + dest] = (_Float16)vrow[i];
}

// ---------------------------------------------------------------------------
// Flash attention per (query block of 128, head). 8 waves, 16 query rows each.
// K/V tiles staged once per workgroup into LDS via async loads.
__global__ __launch_bounds__(256) void attn_kernel(const _Float16* __restrict__ qh,
                                                   const _Float16* __restrict__ kh,
                                                   const _Float16* __restrict__ vt,
                                                   _Float16* __restrict__ og) {
    const int qb = blockIdx.x;          // 0..23
    const int h  = blockIdx.y;          // 0..11
    const int ti = qb / 6;
    const int tid = threadIdx.x;
    const int wave = tid >> 5, lane = tid & 31;
    const int half_ = lane >> 4, l16 = lane & 15;

    __shared__ _Float16 Ks[64][136];      // 64 keys x 128 dims (pad 8)
    __shared__ _Float16 Vs[128][72];      // 128 dims x 64 keys (pad 8)
    __shared__ _Float16 Pls[8][64][17];   // per-wave P tiles, [key][row] padded

    // Q fragments: 4 chunks of K=32 over head dim
    Frag qf[4];
    const int q0 = qb * 128 + wave * 16;
    const _Float16* qbase = qh + (size_t)(q0 + l16) * DIM + h * HD;
    #pragma unroll
    for (int kc = 0; kc < 4; ++kc) {
        const _Float16* p = qbase + kc * 32 + half_ * 16;
        qf[kc].q4[0] = *(const uint4*)p;
        qf[kc].q4[1] = *(const uint4*)(p + 8);
    }

    v8f O[8];
    float mrow[8], lrow[8];
    #pragma unroll
    for (int i = 0; i < 8; ++i) {
        mrow[i] = -3.0e38f;
        lrow[i] = 0.f;
        for (int r = 0; r < 8; ++r) O[i][r] = 0.f;
    }
    const float sc = 0.088388347648318447f;   // 1/sqrt(128)

    for (int kb2 = 0; kb2 < 48; ++kb2) {      // 64-key chunks
        const int tj = kb2 / 12;
        if (!(tj == 0 || (ti - tj <= 1 && tj - ti <= 1))) continue;
        const int kbase = kb2 * 64;

        // Stage K tile (64x128 halfs) and V tile (128x64 halfs) into LDS.
#ifdef ASYNC_LDS
        #pragma unroll
        for (int i = 0; i < 4; ++i) {
            int idx = tid + i * 256;
            int key = idx >> 4, qd = idx & 15;
            __builtin_amdgcn_global_load_async_to_lds_b128(
                GLB_PTR(&kh[(size_t)(kbase + key) * DIM + h * HD + qd * 8]),
                LDS_PTR(&Ks[key][qd * 8]), 0, 0);
        }
        #pragma unroll
        for (int i = 0; i < 4; ++i) {
            int idx = tid + i * 256;
            int dim = idx >> 3, qd = idx & 7;
            __builtin_amdgcn_global_load_async_to_lds_b128(
                GLB_PTR(&vt[(size_t)(h * HD + dim) * SEQ + kbase + qd * 8]),
                LDS_PTR(&Vs[dim][qd * 8]), 0, 0);
        }
        wait_async_lds();
#else
        uint4 tk[4], tv[4];
        #pragma unroll
        for (int i = 0; i < 4; ++i) {
            int idx = tid + i * 256;
            int key = idx >> 4, qd = idx & 15;
            tk[i] = *(const uint4*)&kh[(size_t)(kbase + key) * DIM + h * HD + qd * 8];
            int dim = idx >> 3, qd2 = idx & 7;
            tv[i] = *(const uint4*)&vt[(size_t)(h * HD + dim) * SEQ + kbase + qd2 * 8];
        }
        #pragma unroll
        for (int i = 0; i < 4; ++i) {
            int idx = tid + i * 256;
            int key = idx >> 4, qd = idx & 15;
            *(uint4*)&Ks[key][qd * 8] = tk[i];
            int dim = idx >> 3, qd2 = idx & 7;
            *(uint4*)&Vs[dim][qd2 * 8] = tv[i];
        }
#endif
        __syncthreads();

        // S = Q @ K^T  (4 fragments of 16 keys), B-fragments from LDS
        v8f s[4];
        #pragma unroll
        for (int nf = 0; nf < 4; ++nf) {
            v8f c;
            for (int r = 0; r < 8; ++r) c[r] = 0.f;
            const _Float16* kb_ = &Ks[nf * 16 + l16][half_ * 16];
            #pragma unroll
            for (int kc = 0; kc < 4; ++kc) {
                Frag bf;
                bf.h = *(const v16h*)(kb_ + kc * 32);
                c = __builtin_amdgcn_wmma_f32_16x16x32_f16(false, qf[kc].h, false, bf.h,
                                                           (short)0, c, false, false);
            }
            for (int r = 0; r < 8; ++r) c[r] *= sc;
            s[nf] = c;
        }

        // online softmax (row = r + 8*half_, row spans 16 lanes of the half)
        float mnew[8], scl[8], rs[8];
        #pragma unroll
        for (int r = 0; r < 8; ++r) {
            float mx = s[0][r];
            for (int nf = 1; nf < 4; ++nf) mx = fmaxf(mx, s[nf][r]);
            for (int off = 1; off < 16; off <<= 1) mx = fmaxf(mx, __shfl_xor(mx, off, 32));
            mnew[r] = fmaxf(mrow[r], mx);
            scl[r] = __expf(mrow[r] - mnew[r]);
            mrow[r] = mnew[r];
        }
        #pragma unroll
        for (int nf = 0; nf < 8; ++nf)
            for (int r = 0; r < 8; ++r) O[nf][r] *= scl[r];
        #pragma unroll
        for (int r = 0; r < 8; ++r) rs[r] = 0.f;
        #pragma unroll
        for (int nf = 0; nf < 4; ++nf)
            for (int r = 0; r < 8; ++r) {
                float p = __expf(s[nf][r] - mnew[r]);
                s[nf][r] = p;
                rs[r] += p;
            }
        #pragma unroll
        for (int r = 0; r < 8; ++r) {
            float t = rs[r];
            for (int off = 1; off < 16; off <<= 1) t += __shfl_xor(t, off, 32);
            lrow[r] = lrow[r] * scl[r] + t;
        }

        // P: C-layout -> LDS -> A-layout
        #pragma unroll
        for (int nf = 0; nf < 4; ++nf)
            for (int r = 0; r < 8; ++r)
                Pls[wave][nf * 16 + l16][r + 8 * half_] = (_Float16)s[nf][r];
        __syncthreads();

        // O += P @ V   (V B-fragments from LDS: rows are head dims)
        #pragma unroll
        for (int kc = 0; kc < 2; ++kc) {
            Frag pa;
            #pragma unroll
            for (int j = 0; j < 16; ++j) {
                int k = kc * 32 + half_ * 16 + j;
                pa.e[j] = Pls[wave][k][l16];
            }
            #pragma unroll
            for (int nf2 = 0; nf2 < 8; ++nf2) {
                Frag bv;
                bv.h = *(const v16h*)&Vs[nf2 * 16 + l16][kc * 32 + half_ * 16];
                O[nf2] = __builtin_amdgcn_wmma_f32_16x16x32_f16(false, pa.h, false, bv.h,
                                                                (short)0, O[nf2], false, false);
            }
        }
        __syncthreads();
    }

    // finalize + inverse window partition
    #pragma unroll
    for (int nf2 = 0; nf2 < 8; ++nf2)
        for (int r = 0; r < 8; ++r) {
            int tt = wave * 16 + r + 8 * half_;
            int dst = qb * 128 + tt;
            int b = dst >> 7, tw = dst & 127;
            int bw = b % 3, bh = (b / 3) & 1, bf = b / 6;
            int fi = bf * 8 + (tw >> 4);
            int hi = bh * 4 + ((tw >> 2) & 3);
            int wi = bw * 4 + (tw & 3);
            int orig = fi * 96 + hi * 12 + wi;
            og[(size_t)orig * DIM + h * HD + nf2 * 16 + l16] = (_Float16)(O[nf2][r] / lrow[r]);
        }
}

// ---------------------------------------------------------------------------
extern "C" void kernel_launch(void* const* d_in, const int* in_sizes, int n_in,
                              void* d_out, int out_size, void* d_ws, size_t ws_size,
                              hipStream_t stream) {
    const float* x     = (const float*)d_in[0];
    const float* freqs = (const float*)d_in[1];
    const float* Wq    = (const float*)d_in[2];
    const float* bq    = (const float*)d_in[3];
    const float* Wk    = (const float*)d_in[4];
    const float* bk    = (const float*)d_in[5];
    const float* Wv    = (const float*)d_in[6];
    const float* bv    = (const float*)d_in[7];
    const float* Wo    = (const float*)d_in[8];
    const float* bo    = (const float*)d_in[9];
    const float* gq    = (const float*)d_in[10];
    const float* gk    = (const float*)d_in[11];
    float* out = (float*)d_out;

    char* ws = (char*)d_ws;
    size_t off = 0;
    auto carve = [&](size_t bytes) -> char* {
        char* p = ws + off;
        off += (bytes + 255) & ~(size_t)255;
        return p;
    };
    _Float16* xh  = (_Float16*)carve((size_t)SEQ * DIM * 2);
    _Float16* Wqt = (_Float16*)carve((size_t)DIM * DIM * 2);
    _Float16* Wkt = (_Float16*)carve((size_t)DIM * DIM * 2);
    _Float16* Wvt = (_Float16*)carve((size_t)DIM * DIM * 2);
    _Float16* Wot = (_Float16*)carve((size_t)DIM * DIM * 2);
    float*    qp  = (float*)   carve((size_t)SEQ * DIM * 4);
    float*    kp  = (float*)   carve((size_t)SEQ * DIM * 4);
    float*    vp  = (float*)   carve((size_t)SEQ * DIM * 4);
    _Float16* qhd = (_Float16*)carve((size_t)SEQ * DIM * 2);
    _Float16* khd = (_Float16*)carve((size_t)SEQ * DIM * 2);
    _Float16* vtd = (_Float16*)carve((size_t)SEQ * DIM * 2);
    _Float16* og  = (_Float16*)carve((size_t)SEQ * DIM * 2);

    f32_to_f16_kernel<<<(SEQ * DIM) / 256, 256, 0, stream>>>(x, xh, SEQ * DIM);
    dim3 tg(DIM / 32, DIM / 32);
    transpose_f16_kernel<<<tg, 256, 0, stream>>>(Wq, Wqt);
    transpose_f16_kernel<<<tg, 256, 0, stream>>>(Wk, Wkt);
    transpose_f16_kernel<<<tg, 256, 0, stream>>>(Wv, Wvt);
    transpose_f16_kernel<<<tg, 256, 0, stream>>>(Wo, Wot);

    dim3 gg(DIM / 128, SEQ / 128);
    gemm_bias_f16<<<gg, 256, 0, stream>>>(xh, Wqt, bq, qp, SEQ, DIM, DIM);
    gemm_bias_f16<<<gg, 256, 0, stream>>>(xh, Wkt, bk, kp, SEQ, DIM, DIM);
    gemm_bias_f16<<<gg, 256, 0, stream>>>(xh, Wvt, bv, vp, SEQ, DIM, DIM);

    norm_rope_scatter<<<SEQ, 256, 0, stream>>>(qp, kp, vp, freqs, gq, gk, qhd, khd, vtd);

    attn_kernel<<<dim3(SEQ / 128, NH), 256, 0, stream>>>(qhd, khd, vtd, og);

    gemm_bias_f16<<<gg, 256, 0, stream>>>(og, Wot, bo, out, SEQ, DIM, DIM);
}
